// CFAggregator_63608465654313
// MI455X (gfx1250) — matrix operand order repaired and checked by reading, not compiled, not run on hardware
//
#include <hip/hip_runtime.h>

// ---------------------------------------------------------------------------
// CDNA5 (gfx1250, wave32) implementation of the CF aggregator.
// GEMMs run on v_wmma_f32_16x16x32_bf16 (f32 accumulate); everything
// data-dependent stays fp32 VALU. Problem is gather/bandwidth bound
// (~20 MB traffic @ 23.3 TB/s), so bf16 A/B is plenty for the matrix path.
// ---------------------------------------------------------------------------

typedef __bf16 v16bf __attribute__((ext_vector_type(16)));
typedef float  v8f   __attribute__((ext_vector_type(8)));

#define DIN   128
#define DOUT  128
#define B_    2048
#define MC_   4
#define U_    20000
#define EPB   32          // E / B  (row_idx = repeat(arange(B), 32))
#define RES   0.9f

// ---- wave32 butterfly all-reduce (every lane ends with the sum) -----------
__device__ __forceinline__ float wred(float v) {
#pragma unroll
  for (int o = 16; o > 0; o >>= 1) v += __shfl_xor(v, o, 32);
  return v;
}

__device__ __forceinline__ float elu1(float x) {
  return x > 0.0f ? x : (__expf(x) - 1.0f);
}

// ---------------------------------------------------------------------------
// W [DIN][DOUT] f32 (k-major)  ->  Wt [DOUT][DIN] bf16 (n-major).
// n-major bf16 lets each lane's WMMA B-fragment be one contiguous 32B load.
// ---------------------------------------------------------------------------
__global__ void transpose_w_bf16(const float* __restrict__ W,
                                 __bf16* __restrict__ Wt) {
  int n = blockIdx.x;
  int k = threadIdx.x;
  Wt[n * DIN + k] = (__bf16)W[k * DOUT + n];
}

// ---------------------------------------------------------------------------
// out[r * outStride + 0..127] = table[idx[r], :] @ W      (W given as Wt bf16)
// One wave computes a 16x128 tile: 8 accumulators (16x16 N-tiles),
// K = 128 in 4 steps of 32 -> 32 v_wmma per wave.
//
// A fragment (16-bit 16x32): lane L (g=L>>4, m=L&15) holds row m,
//   element e -> K = kb + 16g + e (contiguous 16, matching B).
// B fragment: lane holds col n = nt*16 + m, K = kb + 16g + e (contiguous 16)
//   -> single aligned 32-byte load from Wt row.
// C/D (32-bit 16x16): acc[nt][rr] is element (row rr + 8g, col nt*16 + m).
// ---------------------------------------------------------------------------
__global__ void gather_gemm_bf16(const float* __restrict__ table,
                                 const int*   __restrict__ idx,
                                 const __bf16* __restrict__ Wt,
                                 float* __restrict__ out,
                                 int nRows, int outStride) {
  const int lane = threadIdx.x & 31;
  const int wave = threadIdx.x >> 5;
  const int tile = blockIdx.x * (blockDim.x >> 5) + wave;
  if (tile * 16 >= nRows) return;            // uniform per wave (EXEC stays ~0)

  const int g = lane >> 4;
  const int m = lane & 15;
  const int r = tile * 16 + m;
  const long gr = (long)idx[r];
  const float* __restrict__ arow = table + gr * DIN;

  v8f acc[8] = {};

#pragma unroll
  for (int kb = 0; kb < DIN; kb += 32) {
    // build bf16 A fragment from gathered f32 row
    v16bf a;
    const int b1 = kb + 16 * g;
    const int b2 = kb + 16 * g + 8;
#pragma unroll
    for (int e = 0; e < 8; ++e) {
      a[e]     = (__bf16)arow[b1 + e];
      a[e + 8] = (__bf16)arow[b2 + e];
    }
#pragma unroll
    for (int nt = 0; nt < 8; ++nt) {
      const __bf16* bp = Wt + (nt * 16 + m) * DIN + kb + 16 * g;
      v16bf bf = *(const v16bf*)bp;          // 32B aligned contiguous load
      acc[nt] = __builtin_amdgcn_wmma_f32_16x16x32_bf16(
          /*neg_a=*/false, a, /*neg_b=*/false, bf,
          /*c_mod=*/(short)0, acc[nt], /*reuse_a=*/false, /*reuse_b=*/false);
    }
  }

#pragma unroll
  for (int nt = 0; nt < 8; ++nt) {
#pragma unroll
    for (int rr = 0; rr < 8; ++rr) {
      const int orow = tile * 16 + rr + 8 * g;
      out[(size_t)orow * outStride + nt * 16 + m] = acc[nt][rr];
    }
  }
}

// ---------------------------------------------------------------------------
// neigh[b][m][d] = mean over edges j of row b with layer m of embedV[col_j][d]
// Edges are row-sorted: row b owns edges [b*EPB, (b+1)*EPB).
// One 128-thread block per row; per-thread register accumulators per layer.
// ---------------------------------------------------------------------------
__global__ void edge_aggregate(const int*   __restrict__ layer_idx,
                               const int*   __restrict__ col_idx,
                               const float* __restrict__ embedV,
                               float* __restrict__ neigh) {
  const int b = blockIdx.x;
  const int d = threadIdx.x;
  float a0 = 0.f, a1 = 0.f, a2 = 0.f, a3 = 0.f;
  int   c0 = 0, c1 = 0, c2 = 0, c3 = 0;
#pragma unroll 4
  for (int j = 0; j < EPB; ++j) {
    const int e  = b * EPB + j;
    const int mm = layer_idx[e];             // broadcast (scalar) load
    const float v = embedV[(size_t)col_idx[e] * DOUT + d];
    if      (mm == 0) { a0 += v; ++c0; }
    else if (mm == 1) { a1 += v; ++c1; }
    else if (mm == 2) { a2 += v; ++c2; }
    else              { a3 += v; ++c3; }
  }
  const size_t base = ((size_t)b * MC_) * DOUT + d;
  neigh[base + 0 * DOUT] = a0 / (float)(c0 > 0 ? c0 : 1);
  neigh[base + 1 * DOUT] = a1 / (float)(c1 > 0 ? c1 : 1);
  neigh[base + 2 * DOUT] = a2 / (float)(c2 > 0 ? c2 : 1);
  neigh[base + 3 * DOUT] = a3 / (float)(c3 > 0 ? c3 : 1);
}

// ---------------------------------------------------------------------------
// Per-row epilogue: persona softmax mixing + 2x2 highway attention + ELU.
// One wave per row b; each lane owns dims {lane, lane+32, lane+64, lane+96}.
// ---------------------------------------------------------------------------
__global__ void finalize(const float* __restrict__ selfA,
                         const float* __restrict__ selfF,
                         const float* __restrict__ neigh,
                         const float* __restrict__ Kb,
                         const float* __restrict__ Qb,
                         const float* __restrict__ mu_w,
                         float* __restrict__ out) {
  const int lane = threadIdx.x & 31;
  const int wave = threadIdx.x >> 5;
  const int b = blockIdx.x * (blockDim.x >> 5) + wave;
  if (b >= B_) return;

  float sa[4], sf[4], ng[MC_][4], kk[2][4], qq[2][4], ms[4], mn[4];
#pragma unroll
  for (int q = 0; q < 4; ++q) {
    const int d = lane + 32 * q;
    sa[q] = selfA[(size_t)b * DOUT + d];
    sf[q] = selfF[(size_t)b * DOUT + d];
    ms[q] = mu_w[d];
    mn[q] = mu_w[DOUT + d];
#pragma unroll
    for (int mm = 0; mm < MC_; ++mm)
      ng[mm][q] = neigh[((size_t)b * MC_ + mm) * DOUT + d];
#pragma unroll
    for (int i = 0; i < 2; ++i) {
      kk[i][q] = Kb[((size_t)b * 2 + i) * DOUT + d];
      qq[i][q] = Qb[((size_t)b * 2 + i) * DOUT + d];
    }
  }

  // persona logits: z_m = (self·mu_s + neigh_m·mu_n) / max(||[self,neigh_m]||, 1e-12)
  float sa2p = 0.f, sdp = 0.f;
#pragma unroll
  for (int q = 0; q < 4; ++q) { sa2p += sa[q] * sa[q]; sdp += sa[q] * ms[q]; }
  const float sa2  = wred(sa2p);
  const float sdot = wred(sdp);

  float z[MC_];
#pragma unroll
  for (int mm = 0; mm < MC_; ++mm) {
    float n2p = 0.f, ndp = 0.f;
#pragma unroll
    for (int q = 0; q < 4; ++q) {
      n2p += ng[mm][q] * ng[mm][q];
      ndp += ng[mm][q] * mn[q];
    }
    const float n2 = wred(n2p);
    const float nd = wred(ndp);
    z[mm] = (sdot + nd) / fmaxf(sqrtf(sa2 + n2), 1e-12f);
  }
  float zmax = z[0];
#pragma unroll
  for (int mm = 1; mm < MC_; ++mm) zmax = fmaxf(zmax, z[mm]);
  float ez[MC_], zs = 0.f;
#pragma unroll
  for (int mm = 0; mm < MC_; ++mm) { ez[mm] = __expf(z[mm] - zmax); zs += ez[mm]; }
  float coef[MC_];
#pragma unroll
  for (int mm = 0; mm < MC_; ++mm) coef[mm] = ez[mm] / zs;

  // neigh_sum + averaged self_agg  (V row 0)
  float v0[4];
#pragma unroll
  for (int q = 0; q < 4; ++q) {
    float nsum = 0.f;
#pragma unroll
    for (int mm = 0; mm < MC_; ++mm) nsum += coef[mm] * ng[mm][q];
    v0[q] = 0.5f * (sa[q] + nsum);
  }

  // 2x2 highway attention: scores_ij = softmax_j(K_i . Q_j / DOUT)
  float s[2][2];
#pragma unroll
  for (int i = 0; i < 2; ++i)
#pragma unroll
    for (int j = 0; j < 2; ++j) {
      float p = 0.f;
#pragma unroll
      for (int q = 0; q < 4; ++q) p += kk[i][q] * qq[j][q];
      s[i][j] = wred(p) * (1.0f / (float)DOUT);
    }
  float p00, p01, p10, p11;
  {
    float mx0 = fmaxf(s[0][0], s[0][1]);
    float e0 = __expf(s[0][0] - mx0), e1 = __expf(s[0][1] - mx0);
    p00 = e0 / (e0 + e1); p01 = e1 / (e0 + e1);
    float mx1 = fmaxf(s[1][0], s[1][1]);
    float f0 = __expf(s[1][0] - mx1), f1 = __expf(s[1][1] - mx1);
    p10 = f0 / (f0 + f1); p11 = f1 / (f0 + f1);
  }

#pragma unroll
  for (int q = 0; q < 4; ++q) {
    const int d = lane + 32 * q;
    const float n0 = p00 * v0[q] + p01 * sf[q];
    const float n1 = p10 * v0[q] + p11 * sf[q];
    const float oa = RES * v0[q] + (1.0f - RES) * n0;
    const float of = RES * sf[q] + (1.0f - RES) * n1;
    out[(size_t)b * DOUT + d]                       = elu1(oa);
    out[(size_t)B_ * DOUT + (size_t)b * DOUT + d]   = elu1(of);
  }
}

// ---------------------------------------------------------------------------
extern "C" void kernel_launch(void* const* d_in, const int* in_sizes, int n_in,
                              void* d_out, int out_size, void* d_ws, size_t ws_size,
                              hipStream_t stream) {
  const int*   nodes      = (const int*)  d_in[0];
  const int*   unique_ids = (const int*)  d_in[1];
  /* d_in[2] row_idx unused: edges are row-sorted, EPB=32 per row */
  const int*   layer_idx  = (const int*)  d_in[3];
  const int*   col_idx    = (const int*)  d_in[4];
  const float* agg_table  = (const float*)d_in[5];
  const float* ff_table   = (const float*)d_in[6];
  const float* Wv_agg     = (const float*)d_in[7];
  const float* Wv_ff      = (const float*)d_in[8];
  const float* Wk         = (const float*)d_in[9];
  const float* Wq         = (const float*)d_in[10];
  const float* mu_w       = (const float*)d_in[11];
  float* out = (float*)d_out;

  // ---- workspace carve-up (~20.5 MB) ----
  char* ws = (char*)d_ws;
  __bf16* WtA = (__bf16*)ws; ws += (size_t)DIN * DOUT * sizeof(__bf16);
  __bf16* WtF = (__bf16*)ws; ws += (size_t)DIN * DOUT * sizeof(__bf16);
  __bf16* WtK = (__bf16*)ws; ws += (size_t)DIN * DOUT * sizeof(__bf16);
  __bf16* WtQ = (__bf16*)ws; ws += (size_t)DIN * DOUT * sizeof(__bf16);
  float* embedV = (float*)ws; ws += (size_t)U_ * DOUT * sizeof(float);
  float* selfA  = (float*)ws; ws += (size_t)B_ * DOUT * sizeof(float);
  float* selfF  = (float*)ws; ws += (size_t)B_ * DOUT * sizeof(float);
  float* neigh  = (float*)ws; ws += (size_t)B_ * MC_ * DOUT * sizeof(float);
  float* Kbuf   = (float*)ws; ws += (size_t)B_ * 2 * DOUT * sizeof(float);
  float* Qbuf   = (float*)ws; ws += (size_t)B_ * 2 * DOUT * sizeof(float);

  // 1) weights -> bf16, n-major
  transpose_w_bf16<<<DOUT, DIN, 0, stream>>>(Wv_agg, WtA);
  transpose_w_bf16<<<DOUT, DIN, 0, stream>>>(Wv_ff,  WtF);
  transpose_w_bf16<<<DOUT, DIN, 0, stream>>>(Wk,     WtK);
  transpose_w_bf16<<<DOUT, DIN, 0, stream>>>(Wq,     WtQ);

  // 2) seven gather-GEMMs (all row counts are multiples of 16)
  const int WPB = 8;                         // waves (tiles) per 256-thread block
  auto blocksFor = [](int rows) { return ((rows / 16) + WPB - 1) / WPB; };
  gather_gemm_bf16<<<blocksFor(U_), WPB * 32, 0, stream>>>(agg_table, unique_ids, WtA, embedV, U_, DOUT);
  gather_gemm_bf16<<<blocksFor(B_), WPB * 32, 0, stream>>>(agg_table, nodes, WtA, selfA, B_, DOUT);
  gather_gemm_bf16<<<blocksFor(B_), WPB * 32, 0, stream>>>(ff_table,  nodes, WtF, selfF, B_, DOUT);
  gather_gemm_bf16<<<blocksFor(B_), WPB * 32, 0, stream>>>(agg_table, nodes, WtK, Kbuf,        B_, 2 * DOUT);
  gather_gemm_bf16<<<blocksFor(B_), WPB * 32, 0, stream>>>(ff_table,  nodes, WtK, Kbuf + DOUT, B_, 2 * DOUT);
  gather_gemm_bf16<<<blocksFor(B_), WPB * 32, 0, stream>>>(agg_table, nodes, WtQ, Qbuf,        B_, 2 * DOUT);
  gather_gemm_bf16<<<blocksFor(B_), WPB * 32, 0, stream>>>(ff_table,  nodes, WtQ, Qbuf + DOUT, B_, 2 * DOUT);

  // 3) segment-mean edge aggregation
  edge_aggregate<<<B_, DOUT, 0, stream>>>(layer_idx, col_idx, embedV, neigh);

  // 4) per-row epilogue (one wave per row, 8 waves per block)
  finalize<<<B_ / 8, 8 * 32, 0, stream>>>(selfA, selfF, neigh, Kbuf, Qbuf, mu_w, out);
}
